// LogicGatedSNN_53111565582354
// MI455X (gfx1250) — matrix operand order repaired and checked by reading, not compiled, not run on hardware
//
#include <hip/hip_runtime.h>

// SNN step for MI455X (gfx1250, wave32).
// GEMV over 256MB of fp32 synapse state -> memory-bound (~11us @ 23.3TB/s).
// Row-sum reduction done on the matrix unit via v_wmma_f32_16x16x32_f16:
//   A = spike vector (0/1) broadcast to all 16 rows (exact in f16)
//   B = (synapse > 50) ? 1 : 0 tile (exact in f16), f32 accumulate.
// Loads: nontemporal b128, full 128B cacheline per row per 32-col tile.

typedef __attribute__((ext_vector_type(16))) _Float16 v16h;
typedef __attribute__((ext_vector_type(8)))  float    v8f;
typedef __attribute__((ext_vector_type(4)))  float    v4f;

#define IN_F   8192
#define OUT_F  8192
#define THRESH 50.0f
#define INHIB  0.5f

// ---------------- Kernel 1: binarized GEMV via WMMA ----------------
// grid = OUT_F/16 blocks, 512 threads (16 waves). Block owns 16 rows.
// Wave w covers K-slice [w*512, w*512+512) as 16 tiles of K=32.
__global__ __launch_bounds__(512) void snn_gemv_wmma(
    const float* __restrict__ spike,   // [IN_F], values in {0,1}
    const float* __restrict__ syn,     // [OUT_F, IN_F]
    float* __restrict__ cur)           // [OUT_F]
{
    __shared__ float partial[16 * 16];             // [wave][n]

    const int tid  = threadIdx.x;
    const int wave = tid >> 5;                     // 0..15
    const int lane = tid & 31;
    const int n    = lane & 15;                    // B column == row-in-block
    const int g    = lane >> 4;                    // lane half-group

    const int row = blockIdx.x * 16 + n;
    const float* __restrict__ rowp = syn + (size_t)row * IN_F;

    v8f c = {};                                    // f32 accumulator tile
    const int kslice = wave * 512;

    for (int t = 0; t < 16; ++t) {
        const int kbase = kslice + t * 32;

        // ---- B(32x16): lane holds 16 contiguous K for its column ----
        // lanes 0-15 -> K=kbase..+15, lanes 16-31 -> K=kbase+16..+31
        const float* bp = rowp + kbase + g * 16;
        v4f b0 = __builtin_nontemporal_load((const v4f*)(bp + 0));
        v4f b1 = __builtin_nontemporal_load((const v4f*)(bp + 4));
        v4f b2 = __builtin_nontemporal_load((const v4f*)(bp + 8));
        v4f b3 = __builtin_nontemporal_load((const v4f*)(bp + 12));
        v16h b;
#pragma unroll
        for (int j = 0; j < 4; ++j) {
            b[j]      = (b0[j] > THRESH) ? (_Float16)1.0f : (_Float16)0.0f;
            b[j + 4]  = (b1[j] > THRESH) ? (_Float16)1.0f : (_Float16)0.0f;
            b[j + 8]  = (b2[j] > THRESH) ? (_Float16)1.0f : (_Float16)0.0f;
            b[j + 12] = (b3[j] > THRESH) ? (_Float16)1.0f : (_Float16)0.0f;
        }

        // ---- A(16x32): spike chunk broadcast to every row M ----
        // lanes 0-15 : K = kbase+{0..15}
        // lanes 16-31: K = kbase+{16..31}
        const float* ap0 = spike + kbase + g * 16;
        const float* ap1 = spike + kbase + g * 16 + 8;
        v4f a0 = *(const v4f*)(ap0 + 0);
        v4f a1 = *(const v4f*)(ap0 + 4);
        v4f a2 = *(const v4f*)(ap1 + 0);
        v4f a3 = *(const v4f*)(ap1 + 4);
        v16h a;
#pragma unroll
        for (int j = 0; j < 4; ++j) {
            a[j]      = (_Float16)a0[j];
            a[j + 4]  = (_Float16)a1[j];
            a[j + 8]  = (_Float16)a2[j];
            a[j + 12] = (_Float16)a3[j];
        }

        // D = A*B + C   (EXEC all-1s here: no divergence in this loop)
        c = __builtin_amdgcn_wmma_f32_16x16x32_f16(
                /*neg_a=*/false, a, /*neg_b=*/false, b,
                /*c_mod=*/(short)0, c, /*reuse_a=*/false, /*reuse_b=*/false);
    }

    // D rows are identical (A rows identical): lane n<16, c[0] = current_partial[row n]
    if (g == 0) partial[wave * 16 + n] = c[0];
    __syncthreads();

    if (tid < 16) {
        float s = 0.0f;
#pragma unroll
        for (int w = 0; w < 16; ++w) s += partial[w * 16 + tid];
        cur[blockIdx.x * 16 + tid] = s;
    }
}

// ---------------- Kernel 2: spikes + global inhibition + state update ----------------
// Single 1024-thread block; each thread owns 8 neurons (coalesced stride-1024).
__global__ __launch_bounds__(1024) void snn_update(
    const float* __restrict__ cur,
    const float* __restrict__ v_mem,
    const float* __restrict__ v_th,
    const float* __restrict__ noise,
    float* __restrict__ out)           // [3*OUT_F] = spikes | v_mem_new | v_th_new
{
    __shared__ float red[1024];
    const int tid = threadIdx.x;

    float spk[8], c8[8], vm8[8], vt8[8];
    float cnt = 0.0f;
#pragma unroll
    for (int k = 0; k < 8; ++k) {
        const int i = tid + k * 1024;
        const float c  = cur[i];
        const float vm = v_mem[i];
        const float vt = v_th[i];
        const float ns = noise[i];
        const float s  = (vm + c + ns >= vt) ? 1.0f : 0.0f;
        spk[k] = s; c8[k] = c; vm8[k] = vm; vt8[k] = vt;
        cnt += s;
        out[i] = s;                                   // spikes
    }

    red[tid] = cnt;
    __syncthreads();
    for (int s = 512; s > 0; s >>= 1) {
        if (tid < s) red[tid] += red[tid + s];
        __syncthreads();
    }
    const float inhib = red[0] * INHIB;

#pragma unroll
    for (int k = 0; k < 8; ++k) {
        const int i = tid + k * 1024;
        const float reset = 1.0f - spk[k];
        const float vnew  = (vm8[k] - inhib + c8[k]) * reset * 0.5f;
        float tnew = vt8[k] + (spk[k] - 0.1f) * 0.01f;
        tnew = fminf(fmaxf(tnew, 0.2f), 5.0f);
        out[OUT_F + i]     = vnew;                    // v_mem_new
        out[2 * OUT_F + i] = tnew;                    // v_th_new
    }
}

extern "C" void kernel_launch(void* const* d_in, const int* in_sizes, int n_in,
                              void* d_out, int out_size, void* d_ws, size_t ws_size,
                              hipStream_t stream) {
    const float* spike = (const float*)d_in[0];   // [1, IN_F]
    const float* syn   = (const float*)d_in[1];   // [OUT_F, IN_F]
    const float* v_mem = (const float*)d_in[2];   // [OUT_F]
    const float* v_th  = (const float*)d_in[3];   // [OUT_F]
    const float* noise = (const float*)d_in[4];   // [OUT_F]
    float* out = (float*)d_out;                   // [3*OUT_F]
    float* cur = (float*)d_ws;                    // [OUT_F] scratch (fully overwritten)

    snn_gemv_wmma<<<OUT_F / 16, 512, 0, stream>>>(spike, syn, cur);
    snn_update<<<1, 1024, 0, stream>>>(cur, v_mem, v_th, noise, out);
}
